// SSMamba_86328842650187
// MI455X (gfx1250) — compile-verified, matrix operand order checked by reading, not run on hardware
//
#include <hip/hip_runtime.h>
#include <hip/hip_bf16.h>

// ---------------- problem constants (match reference) ----------------
#define BB        4
#define LL        2048
#define DM        256          // D_MODEL
#define DS        16           // D_STATE
#define DC        4            // D_CONV
#define DI        512          // D_INNER
#define DTR       17           // DT_RANK
#define NGRP      4
#define GN_EPS    1e-5f
#define MTOT      (BB*LL)      // 8192 rows
#define DBC_LD    64           // padded leading dim for dbc (49 -> 64)

// LDS panel row pitches (keep 16B alignment of async B128 chunks, spread banks)
#define PAD_F32   36           // floats per row for 32-float K panels
#define PAD_F16   40           // halfwords per row for 32-halfword K panels

typedef __attribute__((ext_vector_type(16))) _Float16 v16h;
typedef __attribute__((ext_vector_type(8)))  float    v8f;

// ---------------- CDNA5 async copy (global -> LDS), ASYNCcnt tracked ----------------
__device__ __forceinline__ void async_load_b128(uint32_t lds_off, const void* gaddr) {
  asm volatile("global_load_async_to_lds_b128 %0, %1, off"
               :: "v"(lds_off), "v"(gaddr)
               : "memory");
}
__device__ __forceinline__ void wait_async0() {
  asm volatile("s_wait_asynccnt 0x0" ::: "memory");
}
__device__ __forceinline__ uint32_t lds_addr(const void* p) {
  // flat->LDS aperture mapping keeps the wave-relative LDS offset in addr[31:0]
  return (uint32_t)(uintptr_t)p;
}

// ---------------- WMMA fragment helpers (CDNA5 16x16x32 f16) ----------------
// A-matrix 16x32 (MxK) 16-bit layout: lanes 0-15 row M=lane hold K=0..7 / K=16..23;
// lanes 16-31 hold K=8..15 / K=24..31. B mirrored per-column.
__device__ __forceinline__ v16h load_frag_f32(const float* __restrict__ base,
                                              int ld, int k0, int lane) {
  const int rc = lane & 15;
  const int kh = (lane >> 4) << 3;   // 0 or 8
  const float* __restrict__ p = base + rc * ld + k0 + kh;
  v16h f;
#pragma unroll
  for (int j = 0; j < 8; ++j) {
    f[j]     = (_Float16)p[j];
    f[j + 8] = (_Float16)p[j + 16];
  }
  return f;
}

__device__ __forceinline__ v16h load_frag_f32_rows(const float* __restrict__ base,
                                                   int ld, int k0, int lane,
                                                   int row0, int nrows) {
  const int rc = lane & 15;
  const int kh = (lane >> 4) << 3;
  const bool ok = (row0 + rc) < nrows;
  const float* __restrict__ p = base + (size_t)(row0 + rc) * ld + k0 + kh;
  v16h f;
#pragma unroll
  for (int j = 0; j < 8; ++j) {
    f[j]     = ok ? (_Float16)p[j]      : (_Float16)0.0f;
    f[j + 8] = ok ? (_Float16)p[j + 16] : (_Float16)0.0f;
  }
  return f;
}

// fragment loads out of LDS panels (compiler lowers to ds_load_b128 + cvt)
__device__ __forceinline__ v16h frag_lds_f32(const float* pan, int ld, int row0, int lane) {
  const int rc = lane & 15;
  const int kh = (lane >> 4) << 3;
  const float* q = pan + (row0 + rc) * ld + kh;
  v16h f;
#pragma unroll
  for (int j = 0; j < 8; ++j) {
    f[j]     = (_Float16)q[j];
    f[j + 8] = (_Float16)q[j + 16];
  }
  return f;
}
__device__ __forceinline__ v16h frag_lds_f16(const _Float16* pan, int ld, int row0, int lane) {
  const int rc = lane & 15;
  const int kh = (lane >> 4) << 3;
  const _Float16* q = pan + (row0 + rc) * ld + kh;
  v16h f;
#pragma unroll
  for (int j = 0; j < 8; ++j) {
    f[j]     = q[j];
    f[j + 8] = q[j + 16];
  }
  return f;
}

// C/D 16x16 f32: VGPR r, lane<16 -> (M=r, N=lane); lane>=16 -> (M=r+8, N=lane-16)
__device__ __forceinline__ void store_tile(float* __restrict__ C, int ldc,
                                           int lane, v8f acc) {
  const int n  = lane & 15;
  const int mh = (lane >> 4) << 3;
#pragma unroll
  for (int r = 0; r < 8; ++r) C[(size_t)(mh + r) * ldc + n] = acc[r];
}

__device__ __forceinline__ float silu_f(float x) {
  return x / (1.0f + __expf(-x));
}

// ---------------- Kernel 1: in_proj GEMM  xz = x @ W^T ----------------
// Block 256 threads = 8 waves; tile 16(M) x 256(N); double-buffered async
// LDS staging overlaps the DMA for K-step k+1 with the WMMAs of step k.
// Each wave computes two 16x16 tiles, reusing its A fragment.
__device__ __forceinline__ void stage_inproj(float* apan, float* bpan,
                                             const float* __restrict__ X,
                                             const float* __restrict__ W,
                                             int m0, int nblk, int k0, int tid) {
  if (tid < 128) {                           // A: 16 rows x 8 chunks
    const int r = tid >> 3, cc = tid & 7;
    async_load_b128(lds_addr(&apan[r * PAD_F32 + cc * 4]),
                    X + (size_t)(m0 + r) * DM + k0 + cc * 4);
  }
#pragma unroll
  for (int i = 0; i < 8; ++i) {              // B: 256 rows x 8 chunks = 2048
    const int c = tid + i * 256;
    const int n = c >> 3, cc = c & 7;
    async_load_b128(lds_addr(&bpan[n * PAD_F32 + cc * 4]),
                    W + (size_t)(nblk + n) * DM + k0 + cc * 4);
  }
}

__global__ void __launch_bounds__(256) k_in_proj(
    const float* __restrict__ X,   // (MTOT, DM)
    const float* __restrict__ W,   // (2*DI, DM)
    float* __restrict__ XZ) {      // (MTOT, 2*DI)
  __shared__ float apan[2][16  * PAD_F32];
  __shared__ float bpan[2][256 * PAD_F32];
  const int tid  = threadIdx.x;
  const int lane = tid & 31;
  const int wv   = tid >> 5;                 // 0..7, two 16-wide N tiles each
  const int m0   = blockIdx.x * 16;
  const int nblk = blockIdx.y * 256;
  v8f acc0 = {}, acc1 = {};
  const int NK = DM / 32;                    // 8 K-steps

  stage_inproj(apan[0], bpan[0], X, W, m0, nblk, 0, tid);
  wait_async0();
  __syncthreads();

#pragma unroll 2
  for (int ks = 0; ks < NK; ++ks) {
    const int cur = ks & 1;
    if (ks + 1 < NK)                         // prefetch next K panel (other buffer)
      stage_inproj(apan[1 - cur], bpan[1 - cur], X, W, m0, nblk, (ks + 1) * 32, tid);
    v16h a  = frag_lds_f32(apan[cur], PAD_F32, 0,            lane);
    v16h b0 = frag_lds_f32(bpan[cur], PAD_F32, wv * 32,      lane);
    v16h b1 = frag_lds_f32(bpan[cur], PAD_F32, wv * 32 + 16, lane);
    acc0 = __builtin_amdgcn_wmma_f32_16x16x32_f16(false, a, false, b0,
                                                  (short)0, acc0, false, false);
    acc1 = __builtin_amdgcn_wmma_f32_16x16x32_f16(false, a, false, b1,
                                                  (short)0, acc1, false, false);
    if (ks + 1 < NK) {
      wait_async0();
      __syncthreads();
    }
  }
  store_tile(XZ + (size_t)m0 * (2 * DI) + nblk + wv * 32,      2 * DI, lane, acc0);
  store_tile(XZ + (size_t)m0 * (2 * DI) + nblk + wv * 32 + 16, 2 * DI, lane, acc1);
}

// ---------------- Kernel 2: depthwise causal conv + SiLU ----------------
__global__ void k_conv_silu(const float* __restrict__ XZ,
                            const float* __restrict__ CW,   // (DI, DC)
                            const float* __restrict__ CB,   // (DI,)
                            float* __restrict__ XC) {       // (MTOT, DI)
  const int idx = blockIdx.x * blockDim.x + threadIdx.x;
  if (idx >= MTOT * DI) return;
  const int d = idx & (DI - 1);
  const int m = idx >> 9;
  const int l = m & (LL - 1);
  const int b = m >> 11;
  float acc = CB[d];
#pragma unroll
  for (int k = 0; k < DC; ++k) {
    const int ls = l + k - (DC - 1);
    if (ls >= 0)
      acc = fmaf(XZ[((size_t)(b * LL + ls)) * (2 * DI) + d], CW[d * DC + k], acc);
  }
  XC[idx] = silu_f(acc);
}

// ---------------- Kernel 3: x_proj GEMM  dbc = xc @ W^T  (N=49 padded to 64) ----------------
__global__ void k_x_proj(const float* __restrict__ XC,  // (MTOT, DI)
                         const float* __restrict__ W,   // (49, DI)
                         float* __restrict__ DBC) {     // (MTOT, DBC_LD)
  const int lane = threadIdx.x & 31;
  const int wv   = threadIdx.x >> 5;
  const int m0 = blockIdx.x * 16;
  const int n0 = wv * 16;                 // 4 waves cover padded N=64
  const float* Abase = XC + (size_t)m0 * DI;
  v8f acc = {};
#pragma unroll 4
  for (int k0 = 0; k0 < DI; k0 += 32) {
    __builtin_prefetch(Abase + ((lane & 15) * DI) + k0 + 32, 0, 1);
    v16h a = load_frag_f32(Abase, DI, k0, lane);
    v16h b = load_frag_f32_rows(W, DI, k0, lane, n0, DTR + 2 * DS);
    acc = __builtin_amdgcn_wmma_f32_16x16x32_f16(false, a, false, b,
                                                 (short)0, acc, false, false);
  }
  store_tile(DBC + (size_t)m0 * DBC_LD + n0, DBC_LD, lane, acc);
}

// ---------------- Kernel 4: dt_proj (K=17) + softplus ----------------
__global__ void k_delta(const float* __restrict__ DBC,  // (MTOT, DBC_LD)
                        const float* __restrict__ Wdt,  // (DI, DTR)
                        const float* __restrict__ bdt,  // (DI,)
                        float* __restrict__ DELTA) {    // (MTOT, DI)
  const int idx = blockIdx.x * blockDim.x + threadIdx.x;
  if (idx >= MTOT * DI) return;
  const int d = idx & (DI - 1);
  const int m = idx >> 9;
  const float* __restrict__ row = DBC + (size_t)m * DBC_LD;
  float acc = bdt[d];
#pragma unroll
  for (int r = 0; r < DTR; ++r) acc = fmaf(row[r], Wdt[d * DTR + r], acc);
  DELTA[idx] = (acc > 20.0f) ? acc : log1pf(__expf(acc));
}

// ---------------- Kernel 5: selective scan (software-pipelined) ----------------
// lane layout: n = lane&15 within a 16-lane half-wave; shuffle-reduce over n.
// Loads for step l+1 are issued before the dependent exp/fma chain of step l.
__global__ void k_scan(const float* __restrict__ DELTA, // (MTOT, DI)
                       const float* __restrict__ DBC,   // (MTOT, DBC_LD)
                       const float* __restrict__ XC,    // (MTOT, DI) = u
                       const float* __restrict__ A_log, // (DI, DS)
                       const float* __restrict__ Dp,    // (DI,)
                       float* __restrict__ Y) {         // (MTOT, DI)
  const int t = blockIdx.x * blockDim.x + threadIdx.x;  // BB*DI*DS = 32768
  const int n = t & 15;
  const int d = (t >> 4) & (DI - 1);
  const int b = t >> 13;
  const float Aval = -__expf(A_log[d * DS + n]);
  const float Dpd  = Dp[d];
  float h = 0.0f;
  const size_t baseMd  = (size_t)(b * LL) * DI + d;
  const size_t baseDbc = (size_t)(b * LL) * DBC_LD;
  // prologue: load step 0
  float dt = DELTA[baseMd];
  float u  = XC[baseMd];
  float Bn = DBC[baseDbc + DTR + n];
  float Cn = DBC[baseDbc + DTR + DS + n];
  for (int l = 0; l < LL; ++l) {
    const int l1 = (l + 1 < LL) ? (l + 1) : l;
    const size_t o1 = baseMd + (size_t)l1 * DI;
    const size_t od = baseDbc + (size_t)l1 * DBC_LD;
    const float dtn = DELTA[o1];
    const float un  = XC[o1];
    const float Bnn = DBC[od + DTR + n];
    const float Cnn = DBC[od + DTR + DS + n];
    // dependent chain for step l
    const float dA = __expf(dt * Aval);
    h = fmaf(dA, h, dt * Bn * u);
    float part = h * Cn;
    part += __shfl_xor(part, 1, 32);
    part += __shfl_xor(part, 2, 32);
    part += __shfl_xor(part, 4, 32);
    part += __shfl_xor(part, 8, 32);
    if (n == 0) Y[baseMd + (size_t)l * DI] = part + u * Dpd;
    dt = dtn; u = un; Bn = Bnn; Cn = Cnn;
  }
}

// ---------------- Kernel 6: y *= silu(z), convert to f16 for out_proj ----------------
__global__ void k_gate(const float* __restrict__ Y,
                       const float* __restrict__ XZ,
                       _Float16* __restrict__ YG) {
  const int idx = blockIdx.x * blockDim.x + threadIdx.x;
  if (idx >= MTOT * DI) return;
  const int d = idx & (DI - 1);
  const int m = idx >> 9;
  const float z = XZ[(size_t)m * (2 * DI) + DI + d];
  YG[idx] = (_Float16)(Y[idx] * silu_f(z));
}

// ---------------- Kernel 7: out_proj GEMM  yo = yg @ W^T ----------------
// Block 256 threads = 8 waves; tile 16(M) x 256(N); double-buffered async.
__device__ __forceinline__ void stage_outproj(_Float16* apan, float* bpan,
                                              const _Float16* __restrict__ YG,
                                              const float* __restrict__ W,
                                              int m0, int nblk, int k0, int tid) {
  if (tid < 64) {                            // A f16: 16 rows x 4 chunks
    const int r = tid >> 2, cc = tid & 3;
    async_load_b128(lds_addr(&apan[r * PAD_F16 + cc * 8]),
                    YG + (size_t)(m0 + r) * DI + k0 + cc * 8);
  }
#pragma unroll
  for (int i = 0; i < 8; ++i) {              // B f32: 256 rows x 8 chunks
    const int c = tid + i * 256;
    const int n = c >> 3, cc = c & 7;
    async_load_b128(lds_addr(&bpan[n * PAD_F32 + cc * 4]),
                    W + (size_t)(nblk + n) * DI + k0 + cc * 4);
  }
}

__global__ void __launch_bounds__(256) k_out_proj(
    const _Float16* __restrict__ YG, // (MTOT, DI) f16
    const float* __restrict__ W,     // (DM, DI)
    float* __restrict__ YO) {        // (MTOT, DM)
  __shared__ _Float16 apan[2][16  * PAD_F16];
  __shared__ float    bpan[2][256 * PAD_F32];
  const int tid  = threadIdx.x;
  const int lane = tid & 31;
  const int wv   = tid >> 5;
  const int m0   = blockIdx.x * 16;
  const int nblk = blockIdx.y * 256;         // single y-block (N = DM = 256)
  v8f acc0 = {}, acc1 = {};
  const int NK = DI / 32;                    // 16 K-steps

  stage_outproj(apan[0], bpan[0], YG, W, m0, nblk, 0, tid);
  wait_async0();
  __syncthreads();

#pragma unroll 2
  for (int ks = 0; ks < NK; ++ks) {
    const int cur = ks & 1;
    if (ks + 1 < NK)
      stage_outproj(apan[1 - cur], bpan[1 - cur], YG, W, m0, nblk, (ks + 1) * 32, tid);
    v16h a  = frag_lds_f16(apan[cur], PAD_F16, 0,            lane);
    v16h b0 = frag_lds_f32(bpan[cur], PAD_F32, wv * 32,      lane);
    v16h b1 = frag_lds_f32(bpan[cur], PAD_F32, wv * 32 + 16, lane);
    acc0 = __builtin_amdgcn_wmma_f32_16x16x32_f16(false, a, false, b0,
                                                  (short)0, acc0, false, false);
    acc1 = __builtin_amdgcn_wmma_f32_16x16x32_f16(false, a, false, b1,
                                                  (short)0, acc1, false, false);
    if (ks + 1 < NK) {
      wait_async0();
      __syncthreads();
    }
  }
  store_tile(YO + (size_t)m0 * DM + nblk + wv * 32,      DM, lane, acc0);
  store_tile(YO + (size_t)m0 * DM + nblk + wv * 32 + 16, DM, lane, acc1);
}

// ---------------- Kernel 8: groupnorm stats per (b, group) over (L, C/G) ----------------
__global__ void k_gn_stats(const float* __restrict__ YO,  // (MTOT, DM)
                           float* __restrict__ stats) {   // (BB*NGRP, 2)
  const int bg = blockIdx.x;     // 0..15
  const int b = bg >> 2, g = bg & 3;
  __shared__ float ss[256];
  __shared__ float ss2[256];
  float s = 0.0f, s2 = 0.0f;
  const int CPG = DM / NGRP;     // 64
  for (int i = threadIdx.x; i < LL * CPG; i += 256) {
    const int l = i >> 6, c = i & (CPG - 1);
    const float v = YO[(size_t)(b * LL + l) * DM + g * CPG + c];
    s += v; s2 += v * v;
  }
  ss[threadIdx.x] = s; ss2[threadIdx.x] = s2;
  __syncthreads();
  for (int off = 128; off > 0; off >>= 1) {
    if ((int)threadIdx.x < off) {
      ss[threadIdx.x]  += ss[threadIdx.x + off];
      ss2[threadIdx.x] += ss2[threadIdx.x + off];
    }
    __syncthreads();
  }
  if (threadIdx.x == 0) {
    const float inv = 1.0f / (float)(LL * CPG);
    const float mu = ss[0] * inv;
    stats[bg * 2 + 0] = mu;
    stats[bg * 2 + 1] = ss2[0] * inv - mu * mu;
  }
}

// ---------------- Kernel 9: normalize + affine + SiLU + residual ----------------
__global__ void k_gn_final(const float* __restrict__ YO,
                           const float* __restrict__ stats,
                           const float* __restrict__ gw,
                           const float* __restrict__ gb,
                           const float* __restrict__ Xres,
                           float* __restrict__ OUT) {
  const int idx = blockIdx.x * blockDim.x + threadIdx.x;
  if (idx >= MTOT * DM) return;
  const int c = idx & (DM - 1);
  const int m = idx >> 8;
  const int b = m >> 11;
  const int g = c >> 6;
  const float mu  = stats[(b * NGRP + g) * 2 + 0];
  const float var = stats[(b * NGRP + g) * 2 + 1];
  const float xn = (YO[idx] - mu) * rsqrtf(var + GN_EPS) * gw[c] + gb[c];
  OUT[idx] = silu_f(xn) + Xres[idx];
}

// ---------------- launcher ----------------
extern "C" void kernel_launch(void* const* d_in, const int* in_sizes, int n_in,
                              void* d_out, int out_size, void* d_ws, size_t ws_size,
                              hipStream_t stream) {
  const float* x_graph    = (const float*)d_in[0];
  const float* x_res      = (const float*)d_in[1];
  const float* in_proj_w  = (const float*)d_in[2];
  const float* conv_w     = (const float*)d_in[3];
  const float* conv_b     = (const float*)d_in[4];
  const float* x_proj_w   = (const float*)d_in[5];
  const float* dt_proj_w  = (const float*)d_in[6];
  const float* dt_proj_b  = (const float*)d_in[7];
  const float* A_log      = (const float*)d_in[8];
  const float* D_param    = (const float*)d_in[9];
  const float* out_proj_w = (const float*)d_in[10];
  const float* gn_w       = (const float*)d_in[11];
  const float* gn_b       = (const float*)d_in[12];
  float* out = (float*)d_out;

  size_t off = 0;
  auto wsalloc = [&](size_t bytes) -> void* {
    void* p = (void*)((char*)d_ws + off);
    off += (bytes + 255) & ~(size_t)255;
    return p;
  };
  float*    XZ    = (float*)   wsalloc((size_t)MTOT * 2 * DI * sizeof(float));
  float*    XC    = (float*)   wsalloc((size_t)MTOT * DI * sizeof(float));
  float*    DBC   = (float*)   wsalloc((size_t)MTOT * DBC_LD * sizeof(float));
  float*    DELTA = (float*)   wsalloc((size_t)MTOT * DI * sizeof(float));
  float*    Y     = (float*)   wsalloc((size_t)MTOT * DI * sizeof(float));
  _Float16* YG    = (_Float16*)wsalloc((size_t)MTOT * DI * sizeof(_Float16));
  float*    YO    = (float*)   wsalloc((size_t)MTOT * DM * sizeof(float));
  float*    STATS = (float*)   wsalloc((size_t)BB * NGRP * 2 * sizeof(float));

  // 1. xz = x_graph @ in_proj_w^T   (WMMA, double-buffered async-LDS)
  k_in_proj<<<dim3(MTOT / 16, (2 * DI) / 256), 256, 0, stream>>>(x_graph, in_proj_w, XZ);
  // 2. depthwise causal conv + SiLU
  k_conv_silu<<<(MTOT * DI) / 256, 256, 0, stream>>>(XZ, conv_w, conv_b, XC);
  // 3. dbc = xc @ x_proj_w^T   (WMMA, N padded 49->64)
  k_x_proj<<<dim3(MTOT / 16, 1), 128, 0, stream>>>(XC, x_proj_w, DBC);
  // 4. delta = softplus(dt @ dt_proj_w^T + b)
  k_delta<<<(MTOT * DI) / 256, 256, 0, stream>>>(DBC, dt_proj_w, dt_proj_b, DELTA);
  // 5. selective scan
  k_scan<<<(BB * DI * DS) / 256, 256, 0, stream>>>(DELTA, DBC, XC, A_log, D_param, Y);
  // 6. gate: y * silu(z) -> f16
  k_gate<<<(MTOT * DI) / 256, 256, 0, stream>>>(Y, XZ, YG);
  // 7. yo = yg @ out_proj_w^T   (WMMA, double-buffered async-LDS)
  k_out_proj<<<dim3(MTOT / 16, DM / 256), 256, 0, stream>>>(YG, out_proj_w, YO);
  // 8-9. groupnorm over (L, C/G) per (b,g), SiLU, residual
  k_gn_stats<<<BB * NGRP, 256, 0, stream>>>(YO, STATS);
  k_gn_final<<<(MTOT * DM) / 256, 256, 0, stream>>>(YO, STATS, gn_w, gn_b, x_res, out);
}
